// ContactPointMamba_15676630630611
// MI455X (gfx1250) — matrix operand order repaired and checked by reading, not compiled, not run on hardware
//
#include <hip/hip_runtime.h>

#define BS 4
#define NPTS 4096
#define IN_CH 1156
#define DIM 256
#define DEPTH 8
#define D_INNER 512
#define D_STATE 16
#define D_CONV 4
#define DT_RANK 16
#define MTOT (BS * NPTS)

typedef __attribute__((ext_vector_type(16))) __bf16    v16bf;
typedef __attribute__((ext_vector_type(8)))  float     v8f;
typedef __attribute__((ext_vector_type(4)))  float     f4;
typedef __attribute__((ext_vector_type(2)))  unsigned  u2;

union V16 { v16bf v; unsigned u[8]; };

// pack two floats into one VGPR holding {bf16(y)[31:16], bf16(x)[15:0]} (RNE)
__device__ inline unsigned pack2bf(float x, float y) {
    union { float f; unsigned u; } a, b;
    a.f = x; b.f = y;
    unsigned ua = a.u + (0x7FFFu + ((a.u >> 16) & 1u));
    unsigned ub = b.u + (0x7FFFu + ((b.u >> 16) & 1u));
    return (ua >> 16) | (ub & 0xFFFF0000u);
}

// ---------------------------------------------------------------------------
// A fragment (16-bit 16x32 layout): lane holds K = k0+8*hh..+7 and k0+16+8*hh..+7
// ---------------------------------------------------------------------------
__device__ inline v16bf load_afrag(const float* __restrict__ Arow, int k0, int hh) {
    const float* Ak = Arow + k0 + 8 * hh;
    f4 p0 = *(const f4*)(Ak);
    f4 p1 = *(const f4*)(Ak + 4);
    f4 p2 = *(const f4*)(Ak + 16);
    f4 p3 = *(const f4*)(Ak + 20);
    V16 a;
    a.u[0] = pack2bf(p0[0], p0[1]); a.u[1] = pack2bf(p0[2], p0[3]);
    a.u[2] = pack2bf(p1[0], p1[1]); a.u[3] = pack2bf(p1[2], p1[3]);
    a.u[4] = pack2bf(p2[0], p2[1]); a.u[5] = pack2bf(p2[2], p2[3]);
    a.u[6] = pack2bf(p3[0], p3[1]); a.u[7] = pack2bf(p3[2], p3[3]);
    return a.v;
}

__device__ inline v16bf load_afrag_tail(const float* __restrict__ Arow, int k0, int hh, int K) {
    V16 a;
#pragma unroll
    for (int v = 0; v < 8; ++v) {
        int ka = k0 + ((v < 4) ? (2 * v + 8 * hh) : (16 + 2 * (v - 4) + 8 * hh));
        float x0 = Arow[min(ka, K - 1)];
        float x1 = Arow[min(ka + 1, K - 1)];
        if (ka     >= K) x0 = 0.f;
        if (ka + 1 >= K) x1 = 0.f;
        a.u[v] = pack2bf(x0, x1);
    }
    return a.v;
}

// ---------------------------------------------------------------------------
// B fragment from pre-converted bf16 weights: lane holds K = k0+8*hh..+7 and k0+16+8*hh..+7
// of row (n0+r) of W[N,K]; raw b64 loads, no conversion VALU.
// ---------------------------------------------------------------------------
__device__ inline v16bf load_bfrag16(const unsigned short* __restrict__ Wrow, int k0, int hh) {
    const unsigned short* Wk = Wrow + k0 + 8 * hh;
    u2 q0 = *(const u2*)(Wk);
    u2 q1 = *(const u2*)(Wk + 4);
    u2 q2 = *(const u2*)(Wk + 16);
    u2 q3 = *(const u2*)(Wk + 20);
    V16 b;
    b.u[0] = q0[0]; b.u[1] = q0[1]; b.u[2] = q1[0]; b.u[3] = q1[1];
    b.u[4] = q2[0]; b.u[5] = q2[1]; b.u[6] = q3[0]; b.u[7] = q3[1];
    return b.v;
}

__device__ inline v16bf load_bfrag16_tail(const unsigned short* __restrict__ Wrow,
                                          int k0, int hh, int K) {
    V16 b;
#pragma unroll
    for (int v = 0; v < 8; ++v) {
        int kb = k0 + ((v < 4) ? (2 * v + 8 * hh) : (16 + 2 * (v - 4) + 8 * hh));
        unsigned x0 = Wrow[min(kb, K - 1)];
        unsigned x1 = Wrow[min(kb + 1, K - 1)];
        if (kb     >= K) x0 = 0u;
        if (kb + 1 >= K) x1 = 0u;
        b.u[v] = x0 | (x1 << 16);
    }
    return b.v;
}

__device__ inline void epilogue(float* __restrict__ C, const float* __restrict__ bias,
                                v8f c, int m0, int n0, int N, int r, int hh,
                                int act, int acc) {
    const float bv = bias ? bias[n0 + r] : 0.f;
#pragma unroll
    for (int v = 0; v < 8; ++v) {
        int m = m0 + v + 8 * hh;                 // C/D layout: M = vgpr + 8*half
        float val = c[v] + bv;
        if      (act == 1) val = 0.5f * val * (1.f + erff(val * 0.70710678118f));
        else if (act == 2) val = val / (1.f + __expf(-val));
        else if (act == 3) val = (val > 20.f) ? val : log1pf(__expf(val));
        size_t o = (size_t)m * N + (n0 + r);
        if (acc) C[o] += val; else C[o] = val;
    }
}

#define WMMA_BF16(a, b, c) \
    __builtin_amdgcn_wmma_f32_16x16x32_bf16(false, (a), false, (b), (short)0, (c), false, false)

// NT known at compile time -> branch-free inner loop of NT WMMAs
template <int NT>
__device__ inline void gemm_core(const float* __restrict__ Arow,
                                 const unsigned short* __restrict__ W0,
                                 float* __restrict__ C, const float* __restrict__ bias,
                                 int m0, int nbase, int N, int K, int r, int hh,
                                 int act, int acc)
{
    const unsigned short* __restrict__ W1 = W0 + (size_t)16 * K;
    const unsigned short* __restrict__ W2 = W0 + (size_t)32 * K;
    const unsigned short* __restrict__ W3 = W0 + (size_t)48 * K;
    v8f c0 = {}, c1 = {}, c2 = {}, c3 = {};
    const int K32 = K & ~31;

    for (int k0 = 0; k0 < K32; k0 += 32) {
        __builtin_prefetch(Arow + k0 + 128, 0, 0);
        __builtin_prefetch(W0 + k0 + 256, 0, 0);
        v16bf a = load_afrag(Arow, k0, hh);
        c0 = WMMA_BF16(a, load_bfrag16(W0, k0, hh), c0);
        if (NT > 1) c1 = WMMA_BF16(a, load_bfrag16(W1, k0, hh), c1);
        if (NT > 2) c2 = WMMA_BF16(a, load_bfrag16(W2, k0, hh), c2);
        if (NT > 3) c3 = WMMA_BF16(a, load_bfrag16(W3, k0, hh), c3);
    }
    if (K32 < K) {                                    // ragged-K tail, branch-free loads
        v16bf a = load_afrag_tail(Arow, K32, hh, K);
        c0 = WMMA_BF16(a, load_bfrag16_tail(W0, K32, hh, K), c0);
        if (NT > 1) c1 = WMMA_BF16(a, load_bfrag16_tail(W1, K32, hh, K), c1);
        if (NT > 2) c2 = WMMA_BF16(a, load_bfrag16_tail(W2, K32, hh, K), c2);
        if (NT > 3) c3 = WMMA_BF16(a, load_bfrag16_tail(W3, K32, hh, K), c3);
    }

    epilogue(C, bias, c0, m0, nbase, N, r, hh, act, acc);
    if (NT > 1) epilogue(C, bias, c1, m0, nbase + 16, N, r, hh, act, acc);
    if (NT > 2) epilogue(C, bias, c2, m0, nbase + 32, N, r, hh, act, acc);
    if (NT > 3) epilogue(C, bias, c3, m0, nbase + 48, N, r, hh, act, acc);
}

// ---------------------------------------------------------------------------
// C[M,N] (+)= act(A[M,K](f32) @ W[N,K](bf16)^T + bias)  via v_wmma_f32_16x16x32_bf16
// Each wave: 16 rows x up to 4 N-tiles (A fragment reused). Block = 4 waves.
// act: 0=none 1=gelu 2=silu 3=softplus ; acc: 0=store 1=accumulate
// ---------------------------------------------------------------------------
__global__ void gemm_wmma_bf16(const float* __restrict__ A, const unsigned short* __restrict__ W,
                               const float* __restrict__ bias, float* __restrict__ C,
                               int M, int N, int K, int lda, int act, int acc)
{
    const int wave  = threadIdx.x >> 5;
    const int lane  = threadIdx.x & 31;
    const int r     = lane & 15;
    const int hh    = lane >> 4;
    const int m0    = blockIdx.x * 16;
    const int nbase = blockIdx.y * 256 + wave * 64;
    if (nbase >= N) return;
    int nt = (N - nbase) >> 4;                        // wave-uniform
    if (nt > 4) nt = 4;

    const float* __restrict__ Arow = A + (size_t)(m0 + r) * lda;
    const unsigned short* __restrict__ W0 = W + (size_t)(nbase + r) * K;

    if      (nt >= 4) gemm_core<4>(Arow, W0, C, bias, m0, nbase, N, K, r, hh, act, acc);
    else if (nt == 3) gemm_core<3>(Arow, W0, C, bias, m0, nbase, N, K, r, hh, act, acc);
    else if (nt == 2) gemm_core<2>(Arow, W0, C, bias, m0, nbase, N, K, r, hh, act, acc);
    else              gemm_core<1>(Arow, W0, C, bias, m0, nbase, N, K, r, hh, act, acc);
}

// ---------------------------------------------------------------------------
// f32 -> bf16 (RNE) weight conversion
// ---------------------------------------------------------------------------
__global__ void cvt_bf16_k(const float* __restrict__ in, unsigned short* __restrict__ out,
                           long n)
{
    long i = (long)blockIdx.x * blockDim.x + threadIdx.x;
    if (i >= n) return;
    union { float f; unsigned u; } c; c.f = in[i];
    unsigned u = c.u + (0x7FFFu + ((c.u >> 16) & 1u));
    out[i] = (unsigned short)(u >> 16);
}

// ---------------------------------------------------------------------------
// fusion = concat([x(1), text(512 bcast), time(128 bcast), xyz(3), point_feat(512)])
// ---------------------------------------------------------------------------
__global__ void build_fusion_k(const float* __restrict__ x, const float* __restrict__ pf,
                               const float* __restrict__ lang, const float* __restrict__ tme,
                               const float* __restrict__ xyz, float* __restrict__ out)
{
    size_t i = (size_t)blockIdx.x * blockDim.x + threadIdx.x;
    if (i >= (size_t)MTOT * IN_CH) return;
    int ch = (int)(i % IN_CH);
    size_t p = i / IN_CH;
    int bb = (int)(p / NPTS);
    float v;
    if      (ch < 1)   v = x[p];
    else if (ch < 513) v = lang[(size_t)bb * 512 + (ch - 1)];
    else if (ch < 641) v = tme[(size_t)bb * 128 + (ch - 513)];
    else if (ch < 644) v = xyz[p * 3 + (ch - 641)];
    else               v = pf[p * 512 + (ch - 644)];
    out[i] = v;
}

// ---------------------------------------------------------------------------
// LayerNorm over last dim (256), one row per block of 256 threads
// ---------------------------------------------------------------------------
__global__ void layernorm_k(const float* __restrict__ in, const float* __restrict__ g,
                            const float* __restrict__ b, float* __restrict__ out)
{
    const int row = blockIdx.x, tid = threadIdx.x;
    __shared__ float red[DIM];
    float v = in[(size_t)row * DIM + tid];
    red[tid] = v; __syncthreads();
    for (int s = 128; s > 0; s >>= 1) { if (tid < s) red[tid] += red[tid + s]; __syncthreads(); }
    float mu = red[0] * (1.f / DIM); __syncthreads();
    float d = v - mu;
    red[tid] = d * d; __syncthreads();
    for (int s = 128; s > 0; s >>= 1) { if (tid < s) red[tid] += red[tid + s]; __syncthreads(); }
    float inv = rsqrtf(red[0] * (1.f / DIM) + 1e-5f);
    out[(size_t)row * DIM + tid] = d * inv * g[tid] + b[tid];
}

// ---------------------------------------------------------------------------
// per-batch min/max of xyz
// ---------------------------------------------------------------------------
__global__ void minmax_k(const float* __restrict__ xyz, float* __restrict__ mn,
                         float* __restrict__ mx)
{
    const int bb = blockIdx.x, tid = threadIdx.x;
    __shared__ float smn[3][256], smx[3][256];
    float lmn[3] = {1e30f, 1e30f, 1e30f}, lmx[3] = {-1e30f, -1e30f, -1e30f};
    for (int t = tid; t < NPTS; t += 256)
        for (int d = 0; d < 3; ++d) {
            float v = xyz[((size_t)bb * NPTS + t) * 3 + d];
            lmn[d] = fminf(lmn[d], v); lmx[d] = fmaxf(lmx[d], v);
        }
    for (int d = 0; d < 3; ++d) { smn[d][tid] = lmn[d]; smx[d][tid] = lmx[d]; }
    __syncthreads();
    for (int s = 128; s > 0; s >>= 1) {
        if (tid < s)
            for (int d = 0; d < 3; ++d) {
                smn[d][tid] = fminf(smn[d][tid], smn[d][tid + s]);
                smx[d][tid] = fmaxf(smx[d][tid], smx[d][tid + s]);
            }
        __syncthreads();
    }
    if (tid < 3) { mn[bb * 3 + tid] = smn[tid][0]; mx[bb * 3 + tid] = smx[tid][0]; }
}

// ---------------------------------------------------------------------------
// Hilbert gray-code encode -> 48-bit key (hi<<40 | lo<<16 | t) for stable lexsort
// ---------------------------------------------------------------------------
__global__ void hilbert_k(const float* __restrict__ xyz, const float* __restrict__ mn,
                          const float* __restrict__ mx, unsigned long long* __restrict__ keys)
{
    int i = blockIdx.x * blockDim.x + threadIdx.x;
    if (i >= MTOT) return;
    int bb = i / NPTS, t = i % NPTS;
    int v[3];
    for (int d = 0; d < 3; ++d) {
        float a = xyz[(size_t)i * 3 + d];
        float l = mn[bb * 3 + d], h = mx[bb * 3 + d];
        float nrm = (a - l) / (h - l + 1e-8f) * 0.999f;
        v[d] = (int)(nrm * 65535.0f);
    }
    bool gray[3][16];
    for (int d = 0; d < 3; ++d)
        for (int j = 0; j < 16; ++j) gray[d][j] = (v[d] >> (15 - j)) & 1;
    for (int bit = 0; bit < 15; ++bit)
        for (int dim = 0; dim < 3; ++dim) {
            bool mask = gray[dim][bit];
            if (dim == 0) {
                for (int j = bit + 1; j < 16; ++j) gray[0][j] = gray[0][j] ^ mask;
            } else {
                for (int j = bit + 1; j < 16; ++j) {
                    bool g0 = gray[0][j] ^ mask;
                    bool gd = gray[dim][j];
                    bool fl = (!mask) && (g0 ^ gd);
                    gray[dim][j] = gd ^ fl;
                    gray[0][j]   = g0 ^ fl;
                }
            }
        }
    int bitsum = 0, hi = 0, lo = 0;
    for (int tt = 0; tt < 48; ++tt) {
        bitsum ^= gray[tt % 3][tt / 3] ? 1 : 0;
        if (tt < 24) hi = (hi << 1) | bitsum; else lo = (lo << 1) | bitsum;
    }
    keys[i] = ((unsigned long long)(unsigned)hi << 40) |
              ((unsigned long long)(unsigned)lo << 16) | (unsigned long long)t;
}

// ---------------------------------------------------------------------------
// In-LDS bitonic sort of 4096 keys per batch; emits sort_idx + inverse
// ---------------------------------------------------------------------------
__global__ void sort_k(const unsigned long long* __restrict__ keys,
                       int* __restrict__ sidx, int* __restrict__ uidx)
{
    __shared__ unsigned long long sk[NPTS];
    const int bb = blockIdx.x, tid = threadIdx.x;
    for (int i = tid; i < NPTS; i += blockDim.x) sk[i] = keys[(size_t)bb * NPTS + i];
    __syncthreads();
    for (int k = 2; k <= NPTS; k <<= 1)
        for (int j = k >> 1; j > 0; j >>= 1) {
            for (int i = tid; i < NPTS; i += blockDim.x) {
                int ixj = i ^ j;
                if (ixj > i) {
                    unsigned long long a = sk[i], b = sk[ixj];
                    bool up = ((i & k) == 0);
                    if ((a > b) == up) { sk[i] = b; sk[ixj] = a; }
                }
            }
            __syncthreads();
        }
    for (int i = tid; i < NPTS; i += blockDim.x) {
        int s = (int)(sk[i] & 0xFFFFULL);
        sidx[bb * NPTS + i] = s;
        uidx[bb * NPTS + s] = i;
    }
}

__global__ void gather_k(const float* __restrict__ in, const int* __restrict__ idx,
                         float* __restrict__ out)
{
    size_t i = (size_t)blockIdx.x * blockDim.x + threadIdx.x;
    if (i >= (size_t)MTOT * DIM) return;
    int c = (int)(i % DIM);
    size_t p = i / DIM;
    int bb = (int)(p / NPTS);
    out[i] = in[((size_t)bb * NPTS + idx[p]) * DIM + c];
}

// ---------------------------------------------------------------------------
// depthwise causal conv (kernel 4) + SiLU; dir flips the temporal window
// ---------------------------------------------------------------------------
__global__ void conv_silu_k(const float* __restrict__ xz, const float* __restrict__ w,
                            const float* __restrict__ bc, float* __restrict__ xm, int dir)
{
    size_t i = (size_t)blockIdx.x * blockDim.x + threadIdx.x;
    if (i >= (size_t)MTOT * D_INNER) return;
    int d = (int)(i % D_INNER);
    size_t p = i / D_INNER;
    int bb = (int)(p / NPTS), t = (int)(p % NPTS);
    float acc = bc[d];
#pragma unroll
    for (int j = 0; j < D_CONV; ++j) {
        int tt = dir ? (t + 3 - j) : (t - 3 + j);
        if (tt >= 0 && tt < NPTS)
            acc += w[d * D_CONV + j] * xz[((size_t)bb * NPTS + tt) * (2 * D_INNER) + d];
    }
    xm[i] = acc / (1.f + __expf(-acc));
}

// ---------------------------------------------------------------------------
// Selective scan with ASYNC global->LDS double buffering (CDNA5 path):
// chunk c+1 streams into LDS via global_load_async_to_lds_b128 (ASYNCcnt)
// while the serial recurrence consumes chunk c from the other buffer.
// grid = BS*4 blocks of 128 threads (one d-channel each)
// ---------------------------------------------------------------------------
#define SCAN_CHUNK 128

__device__ inline void scan_issue_chunk(const float* __restrict__ dbl, int bb, int t0,
                                        int dir, float* buf)
{
#pragma unroll
    for (int it = 0; it < 8; ++it) {
        int e4 = (it * 128 + threadIdx.x) * 4;           // float index within chunk
        int s  = e4 >> 5, c = e4 & 31;
        int tg = dir ? (NPTS - 1 - (t0 + s)) : (t0 + s);
        const float* g = dbl + ((size_t)bb * NPTS + tg) * 48 + 16 + c;
        unsigned lds = (unsigned)(size_t)(buf + e4);     // low 32 bits = LDS byte address
        asm volatile("global_load_async_to_lds_b128 %0, %1, off"
                     :: "v"(lds), "v"(g) : "memory");
    }
}

__global__ void scan_k(const float* __restrict__ xm, const float* __restrict__ dt,
                       const float* __restrict__ dbl, const float* __restrict__ A_log,
                       const float* __restrict__ Dp, float* __restrict__ y, int dir)
{
    const int bb = blockIdx.x >> 2;
    const int d  = (blockIdx.x & 3) * 128 + threadIdx.x;
    __shared__ float sBC[2][SCAN_CHUNK * 32];
    float Arow[D_STATE], hst[D_STATE];
#pragma unroll
    for (int n = 0; n < D_STATE; ++n) {
        Arow[n] = -__expf(A_log[d * D_STATE + n]);
        hst[n] = 0.f;
    }
    const float Dd = Dp[d];

    scan_issue_chunk(dbl, bb, 0, dir, sBC[0]);           // prime buffer 0
    int p = 0;
    for (int t0 = 0; t0 < NPTS; t0 += SCAN_CHUNK) {
        asm volatile("s_wait_asynccnt 0x0" ::: "memory"); // my wave's fills done
        __syncthreads();                                  // all waves' fills done
        if (t0 + SCAN_CHUNK < NPTS)
            scan_issue_chunk(dbl, bb, t0 + SCAN_CHUNK, dir, sBC[1 - p]);
        const float* __restrict__ chunk = sBC[p];
        for (int s = 0; s < SCAN_CHUNK; ++s) {
            int tg = dir ? (NPTS - 1 - (t0 + s)) : (t0 + s);
            size_t base = ((size_t)bb * NPTS + tg) * D_INNER + d;
            float dtv = dt[base];
            float uv  = xm[base];
            float du  = dtv * uv;
            const float* Bt = chunk + s * 32;
            const float* Ct = Bt + 16;
            float acc = 0.f;
#pragma unroll
            for (int n = 0; n < D_STATE; ++n) {
                float dA = __expf(dtv * Arow[n]);
                hst[n] = dA * hst[n] + du * Bt[n];
                acc += hst[n] * Ct[n];
            }
            y[base] = acc + uv * Dd;
        }
        p ^= 1;
    }
}

__global__ void gate_k(float* __restrict__ y, const float* __restrict__ xz)
{
    size_t i = (size_t)blockIdx.x * blockDim.x + threadIdx.x;
    if (i >= (size_t)MTOT * D_INNER) return;
    int d = (int)(i % D_INNER);
    size_t p = i / D_INNER;
    float z = xz[p * (2 * D_INNER) + D_INNER + d];
    y[i] *= z / (1.f + __expf(-z));
}

// ---------------------------------------------------------------------------
extern "C" void kernel_launch(void* const* d_in, const int* in_sizes, int n_in,
                              void* d_out, int out_size, void* d_ws, size_t ws_size,
                              hipStream_t stream)
{
    const float* x         = (const float*)d_in[0];
    const float* pf        = (const float*)d_in[1];
    const float* lang      = (const float*)d_in[2];
    const float* tme       = (const float*)d_in[3];
    const float* xyz       = (const float*)d_in[4];
    const float* emb_w1    = (const float*)d_in[5];
    const float* emb_b1    = (const float*)d_in[6];
    const float* emb_w2    = (const float*)d_in[7];
    const float* emb_b2    = (const float*)d_in[8];
    const float* emb_ln_g  = (const float*)d_in[9];
    const float* emb_ln_b  = (const float*)d_in[10];
    const float* ln_g      = (const float*)d_in[11];
    const float* ln_b      = (const float*)d_in[12];
    const float* in_proj_w = (const float*)d_in[13];
    const float* conv_w    = (const float*)d_in[14];
    const float* conv_b    = (const float*)d_in[15];
    const float* x_proj_w  = (const float*)d_in[16];
    const float* dt_proj_w = (const float*)d_in[17];
    const float* dt_proj_b = (const float*)d_in[18];
    const float* A_log     = (const float*)d_in[19];
    const float* Dp        = (const float*)d_in[20];
    const float* out_proj_w= (const float*)d_in[21];
    const float* final_w   = (const float*)d_in[22];
    const float* final_b   = (const float*)d_in[23];

    // ---- workspace carve (float slots) ----
    float* ws    = (float*)d_ws;
    float* xs    = ws;                       // 16384*256        = 4,194,304
    float* arena = ws + 4194304;             // 46,923,776 floats, multi-stage reuse
    float* small = arena + 46923776;
    unsigned long long* keys = (unsigned long long*)small;     // 16384 u64
    int*   sidx = (int*)(small + 32768);
    int*   uidx = sidx + 16384;
    float* mn   = (float*)(uidx + 16384);
    float* mx   = mn + 16;
    // bf16 weight arena (8-byte aligned; counts all multiples of 4)
    unsigned short* wbf      = (unsigned short*)(mx + 16);
    unsigned short* w_inproj = wbf;                      // 8*1024*256 = 2,097,152
    unsigned short* w_xproj  = w_inproj + 2097152;       // 8*48*512   =   196,608
    unsigned short* w_dtproj = w_xproj  + 196608;        // 8*512*16   =    65,536
    unsigned short* w_outprj = w_dtproj + 65536;         // 8*256*512  = 1,048,576
    unsigned short* w_emb1   = w_outprj + 1048576;       // 128*1156   =   147,968
    unsigned short* w_emb2   = w_emb1   + 147968;        // 256*128    =    32,768
    unsigned short* w_final  = w_emb2   + 32768;         // 256*256    =    65,536

    // stage-A overlay
    float* fusion = arena;                   // 16384*1156 = 18,939,904
    float* h1     = arena + 18939904;        // 16384*128  =  2,097,152
    float* hbuf   = arena;                   // reuse fusion space (GEMM2 out)
    float* emb    = arena + 21037056;        // 16384*256
    // layer overlay
    float* h_ln = arena;                     //  4,194,304
    float* xzb  = arena + 4194304;           // 16,777,216
    float* xmb  = arena + 20971520;          //  8,388,608
    float* dblb = arena + 29360128;          //    786,432
    float* dtb  = arena + 30146560;          //  8,388,608
    float* yb   = arena + 38535168;          //  8,388,608

    const int M = MTOT;
    dim3 gblk(128);
    auto ngrid = [](size_t n, int bs) { return (unsigned)((n + bs - 1) / bs); };
    auto ggrid = [](int M_, int N_) { return dim3(M_ / 16, (unsigned)((N_ + 255) / 256)); };
    auto cvt = [&](const float* src, unsigned short* dst, long n) {
        cvt_bf16_k<<<(unsigned)((n + 255) / 256), 256, 0, stream>>>(src, dst, n);
    };

    // ---- Stage 0: one-time (per launch) f32->bf16 weight conversion ----
    cvt(in_proj_w,  w_inproj, 2097152);
    cvt(x_proj_w,   w_xproj,  196608);
    cvt(dt_proj_w,  w_dtproj, 65536);
    cvt(out_proj_w, w_outprj, 1048576);
    cvt(emb_w1,     w_emb1,   147968);
    cvt(emb_w2,     w_emb2,   32768);
    cvt(final_w,    w_final,  65536);

    // ---- Stage A: embedding MLP ----
    build_fusion_k<<<ngrid((size_t)M * IN_CH, 256), 256, 0, stream>>>(x, pf, lang, tme, xyz, fusion);
    gemm_wmma_bf16<<<ggrid(M, 128), gblk, 0, stream>>>(
        fusion, w_emb1, emb_b1, h1, M, 128, IN_CH, IN_CH, /*gelu*/1, 0);
    gemm_wmma_bf16<<<ggrid(M, DIM), gblk, 0, stream>>>(
        h1, w_emb2, emb_b2, hbuf, M, DIM, 128, 128, 0, 0);
    layernorm_k<<<M, DIM, 0, stream>>>(hbuf, emb_ln_g, emb_ln_b, emb);

    // ---- Stage B: Hilbert sort ----
    minmax_k<<<BS, 256, 0, stream>>>(xyz, mn, mx);
    hilbert_k<<<ngrid(M, 256), 256, 0, stream>>>(xyz, mn, mx, keys);
    sort_k<<<BS, 1024, 0, stream>>>(keys, sidx, uidx);
    gather_k<<<ngrid((size_t)M * DIM, 256), 256, 0, stream>>>(emb, sidx, xs);

    // ---- Stage C: 8 bidirectional Mamba layers ----
    for (int i = 0; i < DEPTH; ++i) {
        layernorm_k<<<M, DIM, 0, stream>>>(xs, ln_g + i * DIM, ln_b + i * DIM, h_ln);
        gemm_wmma_bf16<<<ggrid(M, 2 * D_INNER), gblk, 0, stream>>>(
            h_ln, w_inproj + (size_t)i * 2 * D_INNER * DIM, nullptr, xzb,
            M, 2 * D_INNER, DIM, DIM, 0, 0);
        for (int dir = 0; dir < 2; ++dir) {
            conv_silu_k<<<ngrid((size_t)M * D_INNER, 256), 256, 0, stream>>>(
                xzb, conv_w + (size_t)i * D_INNER * D_CONV, conv_b + (size_t)i * D_INNER,
                xmb, dir);
            gemm_wmma_bf16<<<ggrid(M, 48), gblk, 0, stream>>>(
                xmb, w_xproj + (size_t)i * 48 * D_INNER, nullptr, dblb,
                M, 48, D_INNER, D_INNER, 0, 0);
            gemm_wmma_bf16<<<ggrid(M, D_INNER), gblk, 0, stream>>>(
                dblb, w_dtproj + (size_t)i * D_INNER * DT_RANK, dt_proj_b + (size_t)i * D_INNER,
                dtb, M, D_INNER, DT_RANK, 48, /*softplus*/3, 0);
            scan_k<<<BS * 4, 128, 0, stream>>>(
                xmb, dtb, dblb, A_log + (size_t)i * D_INNER * D_STATE, Dp + (size_t)i * D_INNER,
                yb, dir);
            gate_k<<<ngrid((size_t)M * D_INNER, 256), 256, 0, stream>>>(yb, xzb);
            gemm_wmma_bf16<<<ggrid(M, DIM), gblk, 0, stream>>>(
                yb, w_outprj + (size_t)i * DIM * D_INNER, nullptr, xs,
                M, DIM, D_INNER, D_INNER, 0, /*accumulate*/1);
        }
    }

    // ---- Stage D: unsort + final projection ----
    gather_k<<<ngrid((size_t)M * DIM, 256), 256, 0, stream>>>(xs, uidx, h_ln);
    gemm_wmma_bf16<<<ggrid(M, DIM), gblk, 0, stream>>>(
        h_ln, w_final, final_b, (float*)d_out, M, DIM, DIM, DIM, 0, 0);
}
